// GLANT_61942018342918
// MI455X (gfx1250) — matrix operand order repaired and checked by reading, not compiled, not run on hardware
//
#include <hip/hip_runtime.h>
#include <hip/hip_fp16.h>

// Problem constants (from reference)
#define NN 20000
#define FF 256
#define EE 400000
#define HH 4
#define CC 64
#define KK 3
#define HC 256          // H*C
#define NEG_SLOPE 0.2f

typedef __attribute__((ext_vector_type(16))) _Float16 v16h;
typedef __attribute__((ext_vector_type(8)))  _Float16 v8h;
typedef __attribute__((ext_vector_type(8)))  float    v8f;
typedef unsigned int u32x4 __attribute__((ext_vector_type(4)));
typedef int          i32x8 __attribute__((ext_vector_type(8)));
typedef int          i32x4 __attribute__((ext_vector_type(4)));

// LDS layout for staged B tiles: 16 cols x 272 halves (256 + 8-dword TDM pad
// per 128 dwords -> 544B column stride, breaks the 512B bank-conflict stride).
#define BCOL_H 272

// ---------------------------------------------------------------------------
// One-time conversions: x -> f16 (A operand), Wl/Wr -> transposed f16 [N,K]
// so a B-fragment (16 K-values for one output col) is 32 contiguous bytes.
// ---------------------------------------------------------------------------
__global__ __launch_bounds__(256)
void convert_x_f16(const float* __restrict__ x, _Float16* __restrict__ xh)
{
    const int t = blockIdx.x * blockDim.x + threadIdx.x;
    if (t < NN * FF) xh[t] = (_Float16)x[t];
}

__global__ __launch_bounds__(256)
void transpose_w_f16(const float* __restrict__ Wl, const float* __restrict__ Wr,
                     _Float16* __restrict__ WtL, _Float16* __restrict__ WtR)
{
    const int t = blockIdx.x * blockDim.x + threadIdx.x;
    if (t >= FF * HC) return;
    const int n = t >> 8, k = t & 255;          // output: Wt[n*256 + k] = W[k*256 + n]
    WtL[t] = (_Float16)Wl[k * HC + n];
    WtR[t] = (_Float16)Wr[k * HC + n];
}

// ---------------------------------------------------------------------------
// Kernel 1: xl = x@Wl + bl, xr = x@Wr + br via v_wmma_f32_16x16x32_f16.
// Block = 160 threads (5 waves), grid (250, 16): 5*250 = 1250 row tiles exact.
// Wave 0 stages both 8KB B tiles (16 cols, full K) into LDS with the Tensor
// Data Mover (tensor_load_to_lds + s_wait_tensorcnt), with LDS padding of
// 8 dwords per 128 dwords. All 5 waves then run the k-loop from LDS.
// The LDS object is escaped through an input-only inline asm with a memory
// clobber (standing in for the TDM write, which carries no pointer operand);
// reads then go through ldsB directly so they lower to ds_load_b128.
// ---------------------------------------------------------------------------
__device__ __forceinline__ u32x4 tdm_g0(unsigned ldsAddr, const void* gptr)
{
    const unsigned long long ga = (unsigned long long)(size_t)gptr;
    u32x4 g0;
    g0.x = 1u;                                     // count=1, user mode
    g0.y = ldsAddr;                                // lds_addr (bytes)
    g0.z = (unsigned)(ga & 0xffffffffu);           // global_addr[31:0]
    g0.w = (unsigned)((ga >> 32) & 0x01ffffffu)    // global_addr[56:32]
         | (2u << 30);                             // type = 2 ("image")
    return g0;
}

__global__ __launch_bounds__(160)
void proj_wmma(const _Float16* __restrict__ xh,
               const _Float16* __restrict__ WtL, const _Float16* __restrict__ WtR,
               const float* __restrict__ bl, const float* __restrict__ br,
               float* __restrict__ xl, float* __restrict__ xr)
{
    __shared__ _Float16 ldsB[2][16][BCOL_H];       // 17,408 B

    const int lane    = threadIdx.x & 31;
    const int wave    = threadIdx.x >> 5;
    const int rowTile = blockIdx.x * 5 + wave;     // always < 1250
    const int colBase = blockIdx.y * 16;

    if (threadIdx.x < 32) {
        // D# group1: workgroup_mask=0, data_size=1 (2B), pad_enable=1,
        // pad_interval=6 (128 dwords), pad_amount=7 (8 dwords),
        // tensor_dim0=4096, tensor_dim1=1, tile_dim0=4096, tile_dim1=1,
        // tensor_dim0_stride=4096.
        const i32x8 g1 = { (int)0x0F910000, 0x10000000, 0x00010000, 0x10000000,
                           1, 4096, 0, 0 };
        const i32x4 z4 = { 0, 0, 0, 0 };
        const i32x8 z8 = { 0, 0, 0, 0, 0, 0, 0, 0 };
        // ldsB is the only LDS object in this kernel -> offset 0.
        const unsigned lds0 = 0u;
        const unsigned lds1 = (unsigned)(16 * BCOL_H * 2);
        __builtin_amdgcn_tensor_load_to_lds(
            tdm_g0(lds0, WtL + (size_t)colBase * FF), g1, z4, z4, z8, 0);
        __builtin_amdgcn_tensor_load_to_lds(
            tdm_g0(lds1, WtR + (size_t)colBase * FF), g1, z4, z4, z8, 0);
        __builtin_amdgcn_s_wait_tensorcnt(0);
    }
    __syncthreads();

    // Escape ldsB so its contents are not provably-undef (the TDM builtin has
    // no pointer operand for alias analysis). Reads below still go through
    // the ldsB object itself -> addrspace(3) ds_load_b128.
    asm volatile("" :: "r"(&ldsB[0][0][0]) : "memory");

    const int r   = lane & 15;
    const int grp = lane >> 4;
    const _Float16* xrow = xh + (size_t)(rowTile * 16 + r) * FF;
    const int colLocal = lane & 15;
    const int kb       = (lane >> 4) * 16;

    v8f cl = {};
    v8f cr = {};
    for (int k0 = 0; k0 < FF; k0 += 32) {
        const v8h alo = *(const v8h*)(xrow + k0 + grp * 16);
        const v8h ahi = *(const v8h*)(xrow + k0 + grp * 16 + 8);
        const v8h bl0 = *(const v8h*)(&ldsB[0][colLocal][k0 + kb]);
        const v8h bl1 = *(const v8h*)(&ldsB[0][colLocal][k0 + kb + 8]);
        const v8h br0 = *(const v8h*)(&ldsB[1][colLocal][k0 + kb]);
        const v8h br1 = *(const v8h*)(&ldsB[1][colLocal][k0 + kb + 8]);
        v16h a, bLv, bRv;
#pragma unroll
        for (int j = 0; j < 8; ++j) {
            a[j] = alo[j];   a[8 + j] = ahi[j];
            bLv[j] = bl0[j]; bLv[8 + j] = bl1[j];
            bRv[j] = br0[j]; bRv[8 + j] = br1[j];
        }
        cl = __builtin_amdgcn_wmma_f32_16x16x32_f16(false, a, false, bLv,
                                                    (short)0, cl, false, false);
        cr = __builtin_amdgcn_wmma_f32_16x16x32_f16(false, a, false, bRv,
                                                    (short)0, cr, false, false);
    }

    const int n     = lane & 15;
    const int mBase = (lane >> 4) * 8;
#pragma unroll
    for (int rr = 0; rr < 8; ++rr) {
        const int row = rowTile * 16 + mBase + rr;
        const int c   = colBase + n;
        xl[(size_t)row * HC + c] = cl[rr] + bl[c];
        xr[(size_t)row * HC + c] = cr[rr] + br[c];
    }
}

// ---------------------------------------------------------------------------
// Kernel 2: gating logits x@Wg + bg -> softmax -> weights [N,3].
// One wave per node, 8 features per lane, butterfly reduce (wave32).
// ---------------------------------------------------------------------------
__global__ __launch_bounds__(256)
void gate_softmax(const float* __restrict__ x, const float* __restrict__ Wg,
                  const float* __restrict__ bg, float* __restrict__ wts)
{
    const int lane = threadIdx.x & 31;
    const int node = blockIdx.x * 8 + (threadIdx.x >> 5);
    if (node >= NN) return;
    const float* xrow = x + (size_t)node * FF;
    float a0 = 0.f, a1 = 0.f, a2 = 0.f;
#pragma unroll
    for (int j = 0; j < 8; ++j) {
        const int f  = lane * 8 + j;
        const float v = xrow[f];
        a0 += v * Wg[f * KK + 0];
        a1 += v * Wg[f * KK + 1];
        a2 += v * Wg[f * KK + 2];
    }
#pragma unroll
    for (int m = 16; m >= 1; m >>= 1) {
        a0 += __shfl_xor(a0, m, 32);
        a1 += __shfl_xor(a1, m, 32);
        a2 += __shfl_xor(a2, m, 32);
    }
    if (lane == 0) {
        a0 += bg[0]; a1 += bg[1]; a2 += bg[2];
        const float mx = fmaxf(a0, fmaxf(a1, a2));
        const float e0 = __expf(a0 - mx), e1 = __expf(a1 - mx), e2 = __expf(a2 - mx);
        const float inv = 1.0f / (e0 + e1 + e2);
        wts[node * KK + 0] = e0 * inv;
        wts[node * KK + 1] = e1 * inv;
        wts[node * KK + 2] = e2 * inv;
    }
}

// Monotonic float<->uint key for atomicMax-based segment max.
__device__ __forceinline__ unsigned fkey(float f) {
    unsigned u = __float_as_uint(f);
    return (u & 0x80000000u) ? ~u : (u | 0x80000000u);
}
__device__ __forceinline__ float fdec(unsigned k) {
    return (k & 0x80000000u) ? __uint_as_float(k ^ 0x80000000u)
                             : __uint_as_float(~k);
}

// ---------------------------------------------------------------------------
// Kernel 3 (per hop, pass 1): s[e,h] = <att[h], leaky_relu(xl[src]+xr[dst])>,
// store s and atomicMax smax[dst,h]. One wave per edge: lane = 8*h + sub,
// sub covers 8 channels. Hop-0 self loops: e >= nReal -> src=dst=e-nReal.
// ---------------------------------------------------------------------------
__global__ __launch_bounds__(256)
void edge_score_max(const int* __restrict__ src, const int* __restrict__ dst,
                    int nReal, int nTotal,
                    const float* __restrict__ xl, const float* __restrict__ xr,
                    const float* __restrict__ attk,
                    float* __restrict__ sbuf, unsigned* __restrict__ smax)
{
    const int lane = threadIdx.x & 31;
    const int e    = blockIdx.x * 8 + (threadIdx.x >> 5);
    if (e >= nTotal) return;
    int s, d;
    if (e < nReal) { s = src[e]; d = dst[e]; } else { s = d = e - nReal; }

    const int h = lane >> 3, sub = lane & 7;
    const float* pl = xl + (size_t)s * HC + h * CC + sub * 8;
    const float* pr = xr + (size_t)d * HC + h * CC + sub * 8;
    const float* pa = attk + h * CC + sub * 8;
    float acc = 0.f;
#pragma unroll
    for (int j = 0; j < 8; ++j) {
        float v = pl[j] + pr[j];
        v = (v > 0.f) ? v : NEG_SLOPE * v;
        acc += v * pa[j];
    }
    acc += __shfl_xor(acc, 1, 32);
    acc += __shfl_xor(acc, 2, 32);
    acc += __shfl_xor(acc, 4, 32);
    if (sub == 0) {
        sbuf[(size_t)e * HH + h] = acc;
        atomicMax(smax + (size_t)d * HH + h, fkey(acc));
    }
}

// ---------------------------------------------------------------------------
// Kernel 4 (per hop, pass 2): ex = exp(s - smax[dst]);
// denom[dst,h] += ex; num[dst,h,c] += ex * xl[src,h,c].
// ---------------------------------------------------------------------------
__global__ __launch_bounds__(256)
void edge_accum(const int* __restrict__ src, const int* __restrict__ dst,
                int nReal, int nTotal,
                const float* __restrict__ xl,
                const float* __restrict__ sbuf, const unsigned* __restrict__ smax,
                float* __restrict__ denom, float* __restrict__ num)
{
    const int lane = threadIdx.x & 31;
    const int e    = blockIdx.x * 8 + (threadIdx.x >> 5);
    if (e >= nTotal) return;
    int s, d;
    if (e < nReal) { s = src[e]; d = dst[e]; } else { s = d = e - nReal; }

    const int h = lane >> 3, sub = lane & 7;
    const float sv = sbuf[(size_t)e * HH + h];
    const float mx = fdec(smax[(size_t)d * HH + h]);
    const float ex = __expf(sv - mx);
    if (sub == 0) atomicAdd(denom + (size_t)d * HH + h, ex);
    const float* pl = xl + (size_t)s * HC + h * CC + sub * 8;
    float* pn = num + (size_t)d * HC + h * CC + sub * 8;
#pragma unroll
    for (int j = 0; j < 8; ++j)
        atomicAdd(pn + j, ex * pl[j]);
}

// ---------------------------------------------------------------------------
// Kernel 5 (per hop): msgs[n,k,c] = mean_h(num[n,h,c]/denom[n,h]) + bias_out[k,c]
// Empty segments (denom==0) contribute 0 (matches segment_sum of empty set).
// ---------------------------------------------------------------------------
__global__ __launch_bounds__(256)
void hop_finish(const float* __restrict__ num, const float* __restrict__ denom,
                const float* __restrict__ biask, float* __restrict__ msgs, int k)
{
    const int t = blockIdx.x * blockDim.x + threadIdx.x;
    if (t >= NN * CC) return;
    const int n = t / CC, c = t % CC;
    float acc = 0.f;
#pragma unroll
    for (int h = 0; h < HH; ++h) {
        const float dn = denom[(size_t)n * HH + h];
        if (dn > 0.f) acc += num[(size_t)n * HC + h * CC + c] / dn;
    }
    msgs[((size_t)n * KK + k) * CC + c] = acc * (1.0f / HH) + biask[c];
}

// ---------------------------------------------------------------------------
// Kernel 6: out[n,c] = sum_k wts[n,k] * msgs[n,k,c]
// ---------------------------------------------------------------------------
__global__ __launch_bounds__(256)
void final_combine(const float* __restrict__ msgs, const float* __restrict__ wts,
                   float* __restrict__ out)
{
    const int t = blockIdx.x * blockDim.x + threadIdx.x;
    if (t >= NN * CC) return;
    const int n = t / CC, c = t % CC;
    float acc = 0.f;
#pragma unroll
    for (int k = 0; k < KK; ++k)
        acc += wts[n * KK + k] * msgs[((size_t)n * KK + k) * CC + c];
    out[t] = acc;
}

extern "C" void kernel_launch(void* const* d_in, const int* in_sizes, int n_in,
                              void* d_out, int out_size, void* d_ws, size_t ws_size,
                              hipStream_t stream) {
    const float* x    = (const float*)d_in[0];
    const int*   e0   = (const int*)d_in[1];
    const int*   e1   = (const int*)d_in[2];
    const int*   e2   = (const int*)d_in[3];
    const float* Wl   = (const float*)d_in[4];
    const float* bl   = (const float*)d_in[5];
    const float* Wr   = (const float*)d_in[6];
    const float* br   = (const float*)d_in[7];
    const float* att  = (const float*)d_in[8];
    const float* bout = (const float*)d_in[9];
    const float* Wg   = (const float*)d_in[10];
    const float* bg   = (const float*)d_in[11];
    float* out = (float*)d_out;

    // Workspace layout
    char* ws = (char*)d_ws;
    float*    xl    = (float*)ws;    ws += (size_t)NN * HC * 4;        // 20.48 MB
    float*    xr    = (float*)ws;    ws += (size_t)NN * HC * 4;        // 20.48 MB
    float*    wts   = (float*)ws;    ws += (size_t)NN * KK * 4;        // 0.24 MB
    float*    msgs  = (float*)ws;    ws += (size_t)NN * KK * CC * 4;   // 15.36 MB
    float*    sbuf  = (float*)ws;    ws += (size_t)(EE + NN) * HH * 4; // 6.72 MB
    unsigned* smax  = (unsigned*)ws; ws += (size_t)NN * HH * 4;        // 0.32 MB
    float*    denom = (float*)ws;    ws += (size_t)NN * HH * 4;        // 0.32 MB
    float*    num   = (float*)ws;    ws += (size_t)NN * HC * 4;        // 20.48 MB
    _Float16* xh    = (_Float16*)ws; ws += (size_t)NN * FF * 2;        // 10.24 MB
    _Float16* WtL   = (_Float16*)ws; ws += (size_t)FF * HC * 2;        // 0.13 MB
    _Float16* WtR   = (_Float16*)ws; ws += (size_t)FF * HC * 2;        // 0.13 MB

    // One-time precision/layout conversion, then projections (WMMA + TDM)
    convert_x_f16<<<(NN * FF + 255) / 256, 256, 0, stream>>>(x, xh);
    transpose_w_f16<<<(FF * HC + 255) / 256, 256, 0, stream>>>(Wl, Wr, WtL, WtR);
    proj_wmma<<<dim3(250, HC / 16), 160, 0, stream>>>(xh, WtL, WtR, bl, br, xl, xr);
    gate_softmax<<<(NN + 7) / 8, 256, 0, stream>>>(x, Wg, bg, wts);

    const int* srcs[KK] = { e0, e1, e2 };
    for (int k = 0; k < KK; ++k) {
        const int nTotal = (k == 0) ? (EE + NN) : EE;   // hop 0 adds self loops
        const int* src = srcs[k];
        const int* dst = srcs[k] + EE;
        (void)hipMemsetAsync(smax,  0, (size_t)NN * HH * 4, stream);
        (void)hipMemsetAsync(denom, 0, (size_t)NN * HH * 4, stream);
        (void)hipMemsetAsync(num,   0, (size_t)NN * HC * 4, stream);
        const int eb = (nTotal + 7) / 8;
        edge_score_max<<<eb, 256, 0, stream>>>(src, dst, EE, nTotal, xl, xr,
                                               att + (size_t)k * HH * CC, sbuf, smax);
        edge_accum<<<eb, 256, 0, stream>>>(src, dst, EE, nTotal, xl, sbuf, smax,
                                           denom, num);
        hop_finish<<<(NN * CC + 255) / 256, 256, 0, stream>>>(
            num, denom, bout + (size_t)k * CC, msgs, k);
    }
    final_combine<<<(NN * CC + 255) / 256, 256, 0, stream>>>(msgs, wts, out);
}
